// SelfAttention_v1_27341761806416
// MI455X (gfx1250) — matrix operand order repaired and checked by reading, not compiled, not run on hardware
//
#include <hip/hip_runtime.h>
#include <hip/hip_bf16.h>

// ---------------------------------------------------------------------------
// Causal self-attention with column-axis softmax (reference softmaxes axis=1).
// Matmuls: v_wmma_f32_16x16x32_bf16 fed from LDS, with double-buffered
// global_load_async_to_lds_b128 staging (CDNA5 ASYNCcnt path).
//
// Pipeline (workspace-resident, ~200 MB):
//   1. x -> bf16 ; W -> W^T bf16 (LDS-tiled transpose)
//   2. Q/K/V = x @ W          (NT bf16 WMMA GEMM, bf16 out)
//   3. scores = Q @ K^T / 32  (NT bf16 WMMA GEMM, f32 out, causal -inf)
//   4. column max/sum (softmax over query axis), probs -> bf16
//   5. V -> V^T ; context = P @ V^T (NT bf16 WMMA GEMM, f32 -> d_out)
// ---------------------------------------------------------------------------

typedef __attribute__((ext_vector_type(16))) __bf16 bf16x16;
typedef __attribute__((ext_vector_type(8)))  float  f32x8;
typedef __attribute__((ext_vector_type(4)))  int    v4i;

union Frag16 { bf16x16 v; uint4 q[2]; };
union FragC  { f32x8  v; float f[8]; };

// ---- CDNA5 async copy helpers --------------------------------------------

__device__ __forceinline__ void async_copy_b128(void* lptr, const void* gptr)
{
#if __has_builtin(__builtin_amdgcn_global_load_async_to_lds_b128)
    __attribute__((address_space(1))) v4i* ga =
        (__attribute__((address_space(1))) v4i*)const_cast<void*>(gptr);
    __attribute__((address_space(3))) v4i* la =
        (__attribute__((address_space(3))) v4i*)lptr;
    __builtin_amdgcn_global_load_async_to_lds_b128(ga, la, 0, 0);
#else
    unsigned int       l = (unsigned int)(unsigned long long)lptr;
    unsigned long long g = (unsigned long long)gptr;
    asm volatile("global_load_async_to_lds_b128 %0, %1, off"
                 :: "v"(l), "v"(g) : "memory");
#endif
}

__device__ __forceinline__ void wait_async0()
{
#if __has_builtin(__builtin_amdgcn_s_wait_asynccnt)
    __builtin_amdgcn_s_wait_asynccnt(0);
#else
    asm volatile("s_wait_asynccnt 0" ::: "memory");
#endif
}

// ---------------------------------------------------------------------------
// NT GEMM: C[m,n] = scale * sum_k A[m,k] * Bt[n,k]
// A: M x Kd row-major bf16, Bt: N x Kd row-major bf16.
// Block tile 128x128, 8 waves (2x4), wave tile 64x32, K-step 32.
// A/B K-slabs staged into double-buffered LDS via async b128 copies;
// fragments read with ds_load_b128 at the ISA 7.12.2 lane layout.
// LDS row stride = 80 B (64 B data + 16 B pad) to spread banks.
// ---------------------------------------------------------------------------
#define LDS_STRIDE 80
#define LDS_TILE   (128 * LDS_STRIDE)   // 10240 B per tile

template <bool CAUSAL, bool OUT_BF16>
__global__ __launch_bounds__(256)
void gemm_nt_bf16(const __bf16* __restrict__ A, const __bf16* __restrict__ Bt,
                  void* __restrict__ Cv, int M, int N, int Kd,
                  long batchA, long batchB, long batchC, float scale)
{
    __shared__ __align__(16) char lds[2][2][LDS_TILE];   // [buf][A=0/B=1]

    const int lane = threadIdx.x & 31;
    const int wid  = threadIdx.x >> 5;
    const int wm   = wid >> 2;                  // 0..1  -> 64-row slab
    const int wn   = wid & 3;                   // 0..3  -> 32-col slab
    const int m0blk = blockIdx.y * 128;
    const int n0blk = blockIdx.x * 128;
    const int m0    = m0blk + wm * 64;
    const int n0    = n0blk + wn * 32;

    A  += (long)blockIdx.z * batchA;
    Bt += (long)blockIdx.z * batchB;

    const int rsel = lane & 15;                 // row-in-tile selector
    const int hsel = lane >> 4;                 // half-wave selector

    // Stage one 128x32 A slab + 128x32 B slab (1024 x 16B chunks) into buf.
    auto stage = [&](int buf, int k0) {
#pragma unroll
        for (int t = 0; t < 4; ++t) {
            const int c     = (int)threadIdx.x + t * 256;   // 0..1023
            const int which = c >> 9;                       // 0=A, 1=B
            const int idx   = c & 511;
            const int row   = idx >> 2;
            const int ch    = idx & 3;
            const __bf16* g = (which ? (Bt + (long)(n0blk + row) * Kd)
                                     : (A  + (long)(m0blk + row) * Kd))
                              + k0 + ch * 8;
            async_copy_b128(&lds[buf][which][row * LDS_STRIDE + ch * 16], g);
        }
    };

    FragC acc[4][2];
#pragma unroll
    for (int mt = 0; mt < 4; ++mt)
#pragma unroll
        for (int nt = 0; nt < 2; ++nt)
#pragma unroll
            for (int r = 0; r < 8; ++r) acc[mt][nt].f[r] = 0.f;

    const int nsteps = Kd >> 5;
    stage(0, 0);

    for (int s = 0; s < nsteps; ++s) {
        const int buf = s & 1;
        wait_async0();            // staged tile for this iteration landed
        __syncthreads();          // every wave done reading buf^1 & sees data
        if (s + 1 < nsteps) stage(buf ^ 1, (s + 1) << 5);

        const char* la = &lds[buf][0][0];
        const char* lb = &lds[buf][1][0];

        // A fragment (16x32): lane<16 holds K [0..7] & [16..23],
        // lane>=16 holds K [8..15] & [24..31] of row (lane&15).
        Frag16 af[4];
#pragma unroll
        for (int mt = 0; mt < 4; ++mt) {
            const char* p = la + (wm * 64 + mt * 16 + rsel) * LDS_STRIDE + hsel * 16;
            af[mt].q[0] = *(const uint4*)(p);
            af[mt].q[1] = *(const uint4*)(p + 32);
        }
        // B fragment (32x16): lane<16 holds K [0..7] & [16..23] of column (lane&15),
        // lane>=16 holds K [8..15] & [24..31].
        Frag16 bfm[2];
#pragma unroll
        for (int nt = 0; nt < 2; ++nt) {
            const char* p = lb + (wn * 32 + nt * 16 + rsel) * LDS_STRIDE + hsel * 16;
            bfm[nt].q[0] = *(const uint4*)(p);
            bfm[nt].q[1] = *(const uint4*)(p + 32);
        }
#pragma unroll
        for (int mt = 0; mt < 4; ++mt)
#pragma unroll
            for (int nt = 0; nt < 2; ++nt)
                acc[mt][nt].v = __builtin_amdgcn_wmma_f32_16x16x32_bf16(
                    false, af[mt].v, false, bfm[nt].v,
                    (short)0, acc[mt][nt].v, false, false);

        __syncthreads();          // reads of buf done before it is re-staged
    }

    // C/D layout (ISA 7.12.2): VGPR r, lane l -> M = r + 8*(l>>4), N = l&15.
    float*  Cf = (float*)Cv;
    __bf16* Cb = (__bf16*)Cv;
    const long cbase = (long)blockIdx.z * batchC;
#pragma unroll
    for (int mt = 0; mt < 4; ++mt) {
#pragma unroll
        for (int nt = 0; nt < 2; ++nt) {
            const int col = n0 + nt * 16 + rsel;
#pragma unroll
            for (int r = 0; r < 8; ++r) {
                const int row = m0 + mt * 16 + r + hsel * 8;
                float v = acc[mt][nt].f[r] * scale;
                if (CAUSAL && col > row) v = -__builtin_inff();
                const long idx = cbase + (long)row * N + col;
                if (OUT_BF16) Cb[idx] = (__bf16)v;
                else          Cf[idx] = v;
            }
        }
    }
}

// --------------------------- helper kernels --------------------------------

__global__ void f32_to_bf16_kernel(const float* __restrict__ in,
                                   __bf16* __restrict__ out, int n)
{
    int i = blockIdx.x * blockDim.x + threadIdx.x;
    if (i < n) out[i] = (__bf16)in[i];
}

// out[c][r] = (bf16) in[r][c]   (batched; in: R x C, out: C x R)
template <typename Tin>
__global__ void transpose_to_bf16(const Tin* __restrict__ in,
                                  __bf16* __restrict__ out, int R, int C)
{
    __shared__ __bf16 tile[32][33];
    in  += (long)blockIdx.z * R * C;
    out += (long)blockIdx.z * R * C;
    const int c0 = blockIdx.x * 32, r0 = blockIdx.y * 32;
    const int tx = threadIdx.x, ty = threadIdx.y;     // 32 x 8
#pragma unroll
    for (int rr = ty; rr < 32; rr += 8)
        tile[rr][tx] = (__bf16)(float)in[(long)(r0 + rr) * C + c0 + tx];
    __syncthreads();
#pragma unroll
    for (int rr = ty; rr < 32; rr += 8)
        out[(long)(c0 + rr) * R + r0 + tx] = tile[tx][rr];
}

// Per-column (query-axis) softmax statistics: one thread per column j.
__global__ void col_stats_kernel(const float* __restrict__ scores,
                                 float* __restrict__ cmax,
                                 float* __restrict__ csum, int S)
{
    const int j = blockIdx.x * blockDim.x + threadIdx.x;
    const int b = blockIdx.y;
    const float* col = scores + (long)b * S * S + j;
    float m = -__builtin_inff();
    for (int i = 0; i < S; ++i) m = fmaxf(m, col[(long)i * S]);
    float s = 0.f;
    for (int i = 0; i < S; ++i) s += __expf(col[(long)i * S] - m);
    cmax[b * S + j] = m;
    csum[b * S + j] = s;
}

__global__ void make_probs_kernel(const float* __restrict__ scores,
                                  const float* __restrict__ cmax,
                                  const float* __restrict__ csum,
                                  __bf16* __restrict__ attn, int S)
{
    const int idx  = blockIdx.x * blockDim.x + threadIdx.x;
    const int j    = idx % S;
    const int rest = idx / S;
    const int i    = rest % S;
    const int b    = rest / S;
    float v = 0.f;
    if (i >= j)
        v = __expf(scores[idx] - cmax[b * S + j]) / csum[b * S + j];
    attn[idx] = (__bf16)v;
}

// ------------------------------ launcher -----------------------------------

extern "C" void kernel_launch(void* const* d_in, const int* in_sizes, int n_in,
                              void* d_out, int out_size, void* d_ws, size_t ws_size,
                              hipStream_t stream)
{
    (void)in_sizes; (void)n_in; (void)out_size; (void)ws_size;
    const float* x  = (const float*)d_in[0];
    const float* Wq = (const float*)d_in[1];
    const float* Wk = (const float*)d_in[2];
    const float* Wv = (const float*)d_in[3];
    float* out = (float*)d_out;

    const int B = 4, S = 2048, D = 1024;

    size_t off = 0;
    auto alloc = [&](size_t bytes) -> void* {
        void* p = (char*)d_ws + off;
        off += (bytes + 255) & ~(size_t)255;
        return p;
    };
    __bf16* xb   = (__bf16*)alloc((size_t)B * S * D * 2);
    __bf16* Wtq  = (__bf16*)alloc((size_t)D * D * 2);
    __bf16* Wtk  = (__bf16*)alloc((size_t)D * D * 2);
    __bf16* Wtv  = (__bf16*)alloc((size_t)D * D * 2);
    __bf16* Qb   = (__bf16*)alloc((size_t)B * S * D * 2);
    __bf16* Kb   = (__bf16*)alloc((size_t)B * S * D * 2);
    __bf16* Vb   = (__bf16*)alloc((size_t)B * S * D * 2);
    __bf16* Vt   = (__bf16*)alloc((size_t)B * S * D * 2);
    float*  sc   = (float*) alloc((size_t)B * S * S * 4);
    float*  cmax = (float*) alloc((size_t)B * S * 4);
    float*  csum = (float*) alloc((size_t)B * S * 4);
    __bf16* attn = (__bf16*)alloc((size_t)B * S * S * 2);

    const dim3 tb(32, 8);

    // 1. precision conversion + weight transposes
    f32_to_bf16_kernel<<<(B * S * D + 255) / 256, 256, 0, stream>>>(x, xb, B * S * D);
    transpose_to_bf16<float><<<dim3(D / 32, D / 32, 1), tb, 0, stream>>>(Wq, Wtq, D, D);
    transpose_to_bf16<float><<<dim3(D / 32, D / 32, 1), tb, 0, stream>>>(Wk, Wtk, D, D);
    transpose_to_bf16<float><<<dim3(D / 32, D / 32, 1), tb, 0, stream>>>(Wv, Wtv, D, D);

    // 2. Q/K/V projections: (B*S x D) = (B*S x D) @ (D x D)
    const dim3 gqkv(D / 128, (B * S) / 128, 1);
    gemm_nt_bf16<false, true><<<gqkv, 256, 0, stream>>>(xb, Wtq, Qb, B * S, D, D, 0, 0, 0, 1.f);
    gemm_nt_bf16<false, true><<<gqkv, 256, 0, stream>>>(xb, Wtk, Kb, B * S, D, D, 0, 0, 0, 1.f);
    gemm_nt_bf16<false, true><<<gqkv, 256, 0, stream>>>(xb, Wtv, Vb, B * S, D, D, 0, 0, 0, 1.f);

    // 3. scores = Q @ K^T / sqrt(D), causal (-inf where col > row)
    const dim3 gsc(S / 128, S / 128, B);
    gemm_nt_bf16<true, false><<<gsc, 256, 0, stream>>>(
        Qb, Kb, sc, S, S, D, (long)S * D, (long)S * D, (long)S * S, 0.03125f);

    // 4. softmax over query axis (per column), probs -> bf16
    col_stats_kernel<<<dim3(S / 256, B), 256, 0, stream>>>(sc, cmax, csum, S);
    make_probs_kernel<<<(B * S * S) / 256, 256, 0, stream>>>(sc, cmax, csum, attn, S);

    // 5. context = P @ V : transpose V per batch, then NT GEMM -> f32 out
    transpose_to_bf16<__bf16><<<dim3(D / 32, S / 32, B), tb, 0, stream>>>(Vb, Vt, S, D);
    const dim3 gctx(D / 128, S / 128, B);
    gemm_nt_bf16<false, false><<<gctx, 256, 0, stream>>>(
        attn, Vt, out, S, D, S, (long)S * S, (long)S * D, (long)S * D, 1.f);
}
